// MultiHeadAttention_90735479095729
// MI455X (gfx1250) — matrix operand order repaired and checked by reading, not compiled, not run on hardware
//
#include <hip/hip_runtime.h>
#include <hip/hip_bf16.h>

#define NTOK 2048
#define DMODEL 1024
#define NHEAD 16
#define DK 64
#define SM_SCALE 0.125f  // 1/sqrt(64)

typedef __bf16 bf16_t;
typedef __attribute__((ext_vector_type(16))) __bf16 v16bf;
typedef __attribute__((ext_vector_type(8)))  __bf16 v8bf;
typedef __attribute__((ext_vector_type(8)))  float  v8f;
typedef int v4i __attribute__((vector_size(16)));

#define CAT16(lo, hi) __builtin_shufflevector((lo), (hi), 0, 1, 2, 3, 4, 5, 6, 7, \
                                              8, 9, 10, 11, 12, 13, 14, 15)

// ---------------------------------------------------------------------------
// CDNA5 async global->LDS copy (ASYNCcnt-tracked), with portable fallback
// ---------------------------------------------------------------------------
#if __has_builtin(__builtin_amdgcn_global_load_async_to_lds_b128)
#define HAVE_ASYNC_LDS 1
#endif

__device__ __forceinline__ void async_cp16(const bf16_t* g, bf16_t* l) {
#ifdef HAVE_ASYNC_LDS
  __builtin_amdgcn_global_load_async_to_lds_b128(
      (__attribute__((address_space(1))) v4i*)(uintptr_t)g,
      (__attribute__((address_space(3))) v4i*)l,
      /*offset=*/0, /*cpol=*/0);
#else
  *(v8bf*)l = *(const v8bf*)g;  // global_load_b128 + ds_store_b128
#endif
}

__device__ __forceinline__ void wait_async_copies() {
#ifdef HAVE_ASYNC_LDS
#if __has_builtin(__builtin_amdgcn_s_wait_asynccnt)
  __builtin_amdgcn_s_wait_asynccnt(0);
#else
  asm volatile("s_wait_asynccnt 0" ::: "memory");
#endif
#endif
}

// ---------------------------------------------------------------------------
// WMMA helpers (CDNA5 wave32, V_WMMA_F32_16X16X32_BF16)
// ---------------------------------------------------------------------------
__device__ __forceinline__ v8f wmma_bf16(v16bf a, v16bf b, v8f c) {
  return __builtin_amdgcn_wmma_f32_16x16x32_bf16(
      /*neg_a=*/false, a, /*neg_b=*/false, b,
      /*c_mod=*/(short)0, c, /*reuse_a=*/false, /*reuse_b=*/false);
}

// A fragment: 16x32 tile of row-major bf16 matrix (leading dim ld).
__device__ __forceinline__ v16bf load_a_frag(const bf16_t* base, int ld,
                                             int mb, int kb, int lane) {
  int row = mb + (lane & 15);
  int kc  = kb + ((lane >> 4) << 3);
  const bf16_t* p = base + row * ld + kc;
  v8bf lo = *(const v8bf*)(p);
  v8bf hi = *(const v8bf*)(p + 16);
  return CAT16(lo, hi);
}

// B fragment from memory stored "column major for B": element (k,n) at
// base[n*ldn + k]; each lane reads two 8-bf16 runs 16 apart.
__device__ __forceinline__ v16bf load_b_frag(const bf16_t* base, int ldn,
                                             int nb, int kb, int lane) {
  int col = nb + (lane & 15);
  int kr  = kb + ((lane >> 4) << 3);
  const bf16_t* p = base + col * ldn + kr;
  v8bf lo = *(const v8bf*)(p);
  v8bf hi = *(const v8bf*)(p + 16);
  return CAT16(lo, hi);
}

// B fragment from an LDS-staged [64 cols][32 k] bf16 tile.
__device__ __forceinline__ v16bf load_b_lds(const bf16_t* ldsB, int nb, int lane) {
  int col = nb + (lane & 15);
  int kr  = (lane >> 4) << 3;
  const bf16_t* p = ldsB + col * 32 + kr;
  v8bf lo = *(const v8bf*)(p);
  v8bf hi = *(const v8bf*)(p + 16);
  return CAT16(lo, hi);
}

// Stage one 64(col) x 32(k) bf16 B-tile into LDS: 4KB = 256 x 16B chunks,
// 128 threads x 2 chunks.
__device__ __forceinline__ void stage_b_tile(const bf16_t* src, int ld, int kk,
                                             bf16_t* dst, int t) {
#pragma unroll
  for (int u = 0; u < 2; ++u) {
    int c   = t + u * 128;           // chunk id 0..255
    int n   = c >> 2;                // column 0..63
    int off = (c & 3) * 8;           // 0,8,16,24 elements within the 32-k row
    async_cp16(src + n * ld + kk + off, dst + n * 32 + off);
  }
}

template <int M>
__device__ __forceinline__ float swz_xor(float v) {
  int x = __builtin_amdgcn_ds_swizzle(__float_as_int(v), 0x1F | (M << 10));
  return __int_as_float(x);
}

// ---------------------------------------------------------------------------
// Cast / repack kernels (fp32 -> bf16, weight transposes)
// ---------------------------------------------------------------------------
__global__ void k_cast_x(const float* __restrict__ x, bf16_t* __restrict__ xb) {
  int i = blockIdx.x * 256 + threadIdx.x;
  if (i < NTOK * DMODEL) xb[i] = (bf16_t)x[i];
}

// wq (h,d,k) -> wt [h][k_out][d]
__global__ void k_cast_whead(const float* __restrict__ w, bf16_t* __restrict__ wt) {
  int i = blockIdx.x * 256 + threadIdx.x;
  if (i < NHEAD * DK * DMODEL) {
    int d    = i & (DMODEL - 1);
    int rest = i >> 10;
    int h    = rest >> 6;
    int ko   = rest & (DK - 1);
    wt[i] = (bf16_t)w[(h * DMODEL + d) * DK + ko];
  }
}

// wo -> wot[j][i] = wo[i][j]
__global__ void k_cast_wo(const float* __restrict__ wo, bf16_t* __restrict__ wot) {
  int i = blockIdx.x * 256 + threadIdx.x;
  if (i < DMODEL * DMODEL) {
    int col = i & (DMODEL - 1);
    int row = i >> 10;
    wot[i] = (bf16_t)wo[col * DMODEL + row];
  }
}

// ---------------------------------------------------------------------------
// QKV projection. Block = 4 waves, 64 rows x 64 cols per block (one head).
// B tile (weights) double-buffered in LDS via async global->LDS copies and
// shared by all 4 waves; A fragments pipelined in registers. Manual 2x unroll
// keeps a fixed accumulator set (no register rotation copies).
// ---------------------------------------------------------------------------
__global__ void __launch_bounds__(128)
k_qkv(const bf16_t* __restrict__ xb,
      const bf16_t* __restrict__ wqt, const bf16_t* __restrict__ wkt,
      const bf16_t* __restrict__ wvt,
      const float* __restrict__ qbias, const float* __restrict__ kbias,
      const float* __restrict__ vbias,
      bf16_t* __restrict__ q, bf16_t* __restrict__ k, bf16_t* __restrict__ vt) {
  __shared__ bf16_t ldsB[2][64 * 32];
  int t    = threadIdx.x;
  int lane = t & 31;
  int wave = t >> 5;
  int bid  = blockIdx.x;            // which*512 + h*32 + mtb
  int mtb  = bid & 31;
  int h    = (bid >> 5) & 15;
  int which = bid >> 9;             // 0=q 1=k 2=v
  const bf16_t* wt   = (which == 0) ? wqt : (which == 1) ? wkt : wvt;
  const float*  bias = (which == 0) ? qbias : (which == 1) ? kbias : vbias;
  const bf16_t* whd  = wt + h * DK * DMODEL;   // [64][1024]
  int mb = mtb * 64 + wave * 16;

  stage_b_tile(whd, DMODEL, 0, ldsB[0], t);
  wait_async_copies();
  __syncthreads();

  v8f c[4] = {};
  v16bf a0 = load_a_frag(xb, DMODEL, mb, 0, lane);
  v16bf a1;
#pragma unroll 1
  for (int it = 0; it < DMODEL / 32; it += 2) {
    // ---- even step: consume ldsB[0] with a0; stage ldsB[1] + load a1
    stage_b_tile(whd, DMODEL, (it + 1) * 32, ldsB[1], t);
    a1 = load_a_frag(xb, DMODEL, mb, (it + 1) * 32, lane);
    __builtin_prefetch(xb + (mb + (lane & 15)) * DMODEL + it * 32 + 128, 0, 1);
#pragma unroll
    for (int nt = 0; nt < 4; ++nt) {
      v16bf b = load_b_lds(ldsB[0], nt * 16, lane);
      c[nt] = wmma_bf16(a0, b, c[nt]);
    }
    wait_async_copies();
    __syncthreads();

    // ---- odd step: consume ldsB[1] with a1; stage ldsB[0] + load a0
    bool more = (it + 2) < (DMODEL / 32);
    if (more) {
      stage_b_tile(whd, DMODEL, (it + 2) * 32, ldsB[0], t);
      a0 = load_a_frag(xb, DMODEL, mb, (it + 2) * 32, lane);
    }
#pragma unroll
    for (int nt = 0; nt < 4; ++nt) {
      v16bf b = load_b_lds(ldsB[1], nt * 16, lane);
      c[nt] = wmma_bf16(a1, b, c[nt]);
    }
    if (more) {
      wait_async_copies();
      __syncthreads();
    }
  }

  int col0 = lane & 15;
  int hi   = lane >> 4;
#pragma unroll
  for (int nt = 0; nt < 4; ++nt) {
    int   col = nt * 16 + col0;
    float bv  = bias[h * DK + col];
#pragma unroll
    for (int j = 0; j < 8; ++j) {
      int   row = mb + j + 8 * hi;
      float val = c[nt][j] + bv;
      if (which == 2) {
        vt[(h * DK + col) * NTOK + row] = (bf16_t)val;     // V^T [h][d][n]
      } else {
        bf16_t* dst = (which == 0) ? q : k;
        dst[(h * NTOK + row) * DK + col] = (bf16_t)val;    // [h][n][d]
      }
    }
  }
}

// ---------------------------------------------------------------------------
// Flash attention: block = 4 waves; each wave owns 32 queries (2 row-sets) of
// one head. Per 32-key chunk: batched K loads -> 8 WMMAs for S, online
// softmax (ds_swizzle reductions) overlapped with batched V loads, P repacked
// through per-wave LDS (s_wait_dscnt 0), 8 WMMAs for P*V.
// ---------------------------------------------------------------------------
__global__ void __launch_bounds__(128)
k_attn(const bf16_t* __restrict__ q, const bf16_t* __restrict__ k,
       const bf16_t* __restrict__ vt, bf16_t* __restrict__ concat) {
  __shared__ bf16_t plds[4][32 * 32];
  int t    = threadIdx.x;
  int lane = t & 31;
  int wave = t >> 5;
  int bid  = blockIdx.x;          // h*16 + mtg
  int mtg  = bid & 15;
  int h    = bid >> 4;
  int mb   = (mtg * 4 + wave) * 32;   // 32 queries per wave
  const bf16_t* qh  = q  + h * NTOK * DK;
  const bf16_t* kh  = k  + h * NTOK * DK;
  const bf16_t* vth = vt + h * DK * NTOK;
  bf16_t* pw = &plds[wave][0];

  v16bf aq[2][2];
#pragma unroll
  for (int r = 0; r < 2; ++r)
#pragma unroll
    for (int kp = 0; kp < 2; ++kp)
      aq[r][kp] = load_a_frag(qh, DK, mb + r * 16, kp * 32, lane);

  float m[2][8], l[2][8];
  v8f o[2][4] = {};
#pragma unroll
  for (int r = 0; r < 2; ++r)
#pragma unroll
    for (int j = 0; j < 8; ++j) { m[r][j] = -1e30f; l[r][j] = 0.f; }
  int col = lane & 15;
  int hi  = lane >> 4;

  for (int kc = 0; kc < NTOK; kc += 32) {
    // ---- batched K fragment loads, then 8 score WMMAs
    v16bf bk[4];
    bk[0] = load_b_frag(kh, DK, kc,      0,  lane);
    bk[1] = load_b_frag(kh, DK, kc,      32, lane);
    bk[2] = load_b_frag(kh, DK, kc + 16, 0,  lane);
    bk[3] = load_b_frag(kh, DK, kc + 16, 32, lane);
    v8f s[2][2] = {};
#pragma unroll
    for (int r = 0; r < 2; ++r) {
      s[r][0] = wmma_bf16(aq[r][0], bk[0], s[r][0]);
      s[r][0] = wmma_bf16(aq[r][1], bk[1], s[r][0]);
      s[r][1] = wmma_bf16(aq[r][0], bk[2], s[r][1]);
      s[r][1] = wmma_bf16(aq[r][1], bk[3], s[r][1]);
    }

    // ---- start V fragment loads early; they overlap the softmax VALU work
    v16bf bv[4];
#pragma unroll
    for (int nt = 0; nt < 4; ++nt)
      bv[nt] = load_b_frag(vth, NTOK, nt * 16, kc, lane);

    // ---- online softmax per row-set; write P (bf16) into per-wave LDS
#pragma unroll
    for (int r = 0; r < 2; ++r) {
#pragma unroll
      for (int j = 0; j < 8; ++j) {
        float a0 = s[r][0][j] * SM_SCALE;
        float a1 = s[r][1][j] * SM_SCALE;
        float v = fmaxf(a0, a1);
        v = fmaxf(v, swz_xor<1>(v));
        v = fmaxf(v, swz_xor<2>(v));
        v = fmaxf(v, swz_xor<4>(v));
        v = fmaxf(v, swz_xor<8>(v));
        float mn = fmaxf(m[r][j], v);
        float sc = __expf(m[r][j] - mn);
        m[r][j] = mn;
        float e0 = __expf(a0 - mn);
        float e1 = __expf(a1 - mn);
        float rs = e0 + e1;
        rs += swz_xor<1>(rs);
        rs += swz_xor<2>(rs);
        rs += swz_xor<4>(rs);
        rs += swz_xor<8>(rs);
        l[r][j] = l[r][j] * sc + rs;
        o[r][0][j] *= sc; o[r][1][j] *= sc; o[r][2][j] *= sc; o[r][3][j] *= sc;
        int rr = r * 16 + j + 8 * hi;
        pw[rr * 32 + col]      = (bf16_t)e0;
        pw[rr * 32 + col + 16] = (bf16_t)e1;
      }
    }

    // intra-wave LDS ordering: counter wait is the correct sync here
    asm volatile("s_wait_dscnt 0" ::: "memory");

    v16bf pa[2];
#pragma unroll
    for (int r = 0; r < 2; ++r) {
      int row = r * 16 + (lane & 15);
      int kc2 = (lane >> 4) << 3;
      const bf16_t* p = pw + row * 32 + kc2;
      v8bf lo = *(const v8bf*)(p);
      v8bf hh = *(const v8bf*)(p + 16);
      pa[r] = CAT16(lo, hh);
    }

    // ---- O += P * V : 8 WMMAs, V fragments reused by both row-sets
#pragma unroll
    for (int nt = 0; nt < 4; ++nt) {
      o[0][nt] = wmma_bf16(pa[0], bv[nt], o[0][nt]);
      o[1][nt] = wmma_bf16(pa[1], bv[nt], o[1][nt]);
    }
  }

  // Normalize and store concat [n][h*64 + d] as bf16
#pragma unroll
  for (int r = 0; r < 2; ++r)
#pragma unroll
    for (int nt = 0; nt < 4; ++nt)
#pragma unroll
      for (int j = 0; j < 8; ++j) {
        int row = mb + r * 16 + j + 8 * hi;
        concat[row * DMODEL + h * DK + nt * 16 + col] =
            (bf16_t)(o[r][nt][j] / l[r][j]);
      }
}

// ---------------------------------------------------------------------------
// Out projection + bias + residual. Same LDS double-buffered B staging.
// ---------------------------------------------------------------------------
__global__ void __launch_bounds__(128)
k_oproj(const bf16_t* __restrict__ concat, const bf16_t* __restrict__ wot,
        const float* __restrict__ o_bias, const float* __restrict__ x,
        float* __restrict__ out) {
  __shared__ bf16_t ldsB[2][64 * 32];
  int t    = threadIdx.x;
  int lane = t & 31;
  int wave = t >> 5;
  int bid  = blockIdx.x;     // mtb*16 + ntg
  int ntg  = bid & 15;
  int mtb  = bid >> 4;
  int mb   = mtb * 64 + wave * 16;
  int nb0  = ntg * 64;
  const bf16_t* wcols = wot + nb0 * DMODEL;

  stage_b_tile(wcols, DMODEL, 0, ldsB[0], t);
  wait_async_copies();
  __syncthreads();

  v8f c[4] = {};
  v16bf a0 = load_a_frag(concat, DMODEL, mb, 0, lane);
  v16bf a1;
#pragma unroll 1
  for (int it = 0; it < DMODEL / 32; it += 2) {
    stage_b_tile(wcols, DMODEL, (it + 1) * 32, ldsB[1], t);
    a1 = load_a_frag(concat, DMODEL, mb, (it + 1) * 32, lane);
    __builtin_prefetch(concat + (mb + (lane & 15)) * DMODEL + it * 32 + 128, 0, 1);
#pragma unroll
    for (int nt = 0; nt < 4; ++nt) {
      v16bf b = load_b_lds(ldsB[0], nt * 16, lane);
      c[nt] = wmma_bf16(a0, b, c[nt]);
    }
    wait_async_copies();
    __syncthreads();

    bool more = (it + 2) < (DMODEL / 32);
    if (more) {
      stage_b_tile(wcols, DMODEL, (it + 2) * 32, ldsB[0], t);
      a0 = load_a_frag(concat, DMODEL, mb, (it + 2) * 32, lane);
    }
#pragma unroll
    for (int nt = 0; nt < 4; ++nt) {
      v16bf b = load_b_lds(ldsB[1], nt * 16, lane);
      c[nt] = wmma_bf16(a1, b, c[nt]);
    }
    if (more) {
      wait_async_copies();
      __syncthreads();
    }
  }

  int col0 = lane & 15;
  int hi   = lane >> 4;
#pragma unroll
  for (int nt = 0; nt < 4; ++nt) {
    int   col = nb0 + nt * 16 + col0;
    float ob  = o_bias[col];
#pragma unroll
    for (int j = 0; j < 8; ++j) {
      int row = mb + j + 8 * hi;
      out[row * DMODEL + col] = c[nt][j] + ob + x[row * DMODEL + col];
    }
  }
}

// ---------------------------------------------------------------------------
// LayerNorm over each row of out -> final fp32 output
// ---------------------------------------------------------------------------
__global__ void __launch_bounds__(256)
k_ln(const float* __restrict__ out, const float* __restrict__ alpha,
     const float* __restrict__ beta, float* __restrict__ y) {
  __shared__ float red[256];
  int row = blockIdx.x;
  int t   = threadIdx.x;
  const float* r = out + row * DMODEL;

  float s = 0.f;
  for (int i = t; i < DMODEL; i += 256) s += r[i];
  red[t] = s; __syncthreads();
  for (int st = 128; st > 0; st >>= 1) { if (t < st) red[t] += red[t + st]; __syncthreads(); }
  float mu = red[0] / DMODEL;
  __syncthreads();

  float v = 0.f;
  for (int i = t; i < DMODEL; i += 256) { float d = r[i] - mu; v += d * d; }
  red[t] = v; __syncthreads();
  for (int st = 128; st > 0; st >>= 1) { if (t < st) red[t] += red[t + st]; __syncthreads(); }
  float inv = rsqrtf(red[0] / DMODEL + 1e-5f);

  for (int i = t; i < DMODEL; i += 256)
    y[row * DMODEL + i] = alpha[i] * (r[i] - mu) * inv + beta[i];
}

// ---------------------------------------------------------------------------
extern "C" void kernel_launch(void* const* d_in, const int* in_sizes, int n_in,
                              void* d_out, int out_size, void* d_ws, size_t ws_size,
                              hipStream_t stream) {
  const float* x      = (const float*)d_in[0];
  const float* wq     = (const float*)d_in[1];
  const float* qbias  = (const float*)d_in[2];
  const float* wk     = (const float*)d_in[3];
  const float* kbias  = (const float*)d_in[4];
  const float* wv     = (const float*)d_in[5];
  const float* vbias  = (const float*)d_in[6];
  const float* wo     = (const float*)d_in[7];
  const float* obias  = (const float*)d_in[8];
  const float* alpha  = (const float*)d_in[9];
  const float* beta   = (const float*)d_in[10];
  float* y = (float*)d_out;

  char* ws = (char*)d_ws;
  const size_t MB = 1024 * 1024;
  bf16_t* xb     = (bf16_t*)(ws + 0 * MB);   // 4 MB
  bf16_t* wqt    = (bf16_t*)(ws + 4 * MB);   // 2 MB
  bf16_t* wkt    = (bf16_t*)(ws + 6 * MB);   // 2 MB
  bf16_t* wvt    = (bf16_t*)(ws + 8 * MB);   // 2 MB
  bf16_t* wot    = (bf16_t*)(ws + 10 * MB);  // 2 MB
  bf16_t* qbuf   = (bf16_t*)(ws + 12 * MB);  // 4 MB
  bf16_t* kbuf   = (bf16_t*)(ws + 16 * MB);  // 4 MB
  bf16_t* vtbuf  = (bf16_t*)(ws + 20 * MB);  // 4 MB
  bf16_t* concat = (bf16_t*)(ws + 24 * MB);  // 4 MB
  float*  obuf   = (float*) (ws + 28 * MB);  // 8 MB

  k_cast_x<<<(NTOK * DMODEL) / 256, 256, 0, stream>>>(x, xb);
  k_cast_whead<<<(NHEAD * DK * DMODEL) / 256, 256, 0, stream>>>(wq, wqt);
  k_cast_whead<<<(NHEAD * DK * DMODEL) / 256, 256, 0, stream>>>(wk, wkt);
  k_cast_whead<<<(NHEAD * DK * DMODEL) / 256, 256, 0, stream>>>(wv, wvt);
  k_cast_wo<<<(DMODEL * DMODEL) / 256, 256, 0, stream>>>(wo, wot);

  k_qkv<<<3 * NHEAD * (NTOK / 64), 128, 0, stream>>>(
      xb, wqt, wkt, wvt, qbias, kbias, vbias, qbuf, kbuf, vtbuf);

  k_attn<<<NHEAD * (NTOK / 128), 128, 0, stream>>>(qbuf, kbuf, vtbuf, concat);

  k_oproj<<<(NTOK / 64) * (DMODEL / 64), 128, 0, stream>>>(concat, wot, obias, x, obuf);

  k_ln<<<NTOK, 256, 0, stream>>>(obuf, alpha, beta, y);
}
